// TauEndToEndSimple_23665269801355
// MI455X (gfx1250) — compile-verified, matrix operand order checked by reading, not run on hardware
//
#include <hip/hip_runtime.h>
#include <hip/hip_bf16.h>
#include <math.h>

typedef __attribute__((ext_vector_type(16))) _Float16 v16h;
typedef __attribute__((ext_vector_type(8)))  float    v8f;
typedef unsigned int u32x4 __attribute__((ext_vector_type(4)));
typedef int          i32x4 __attribute__((ext_vector_type(4)));
typedef int          i32x8 __attribute__((ext_vector_type(8)));

union AF { v16h v; _Float16 h[16]; };
union CF { v8f  v; float    f[8];  };

__device__ __forceinline__ float elu_f(float x) { return x > 0.f ? x : (expf(x) - 1.f); }

#if __has_builtin(__builtin_amdgcn_tensor_load_to_lds)
#define HAVE_TDM 1
#else
#define HAVE_TDM 0
#endif

#if HAVE_TDM
// 1-D contiguous f16 block: global -> LDS via Tensor Data Mover.
// D# per CDNA5 ISA: group0 {count=1, lds_addr, global_addr, type=2},
// group1 {data_size=2B, tensor_dim0=tile_dim0=nelem, dim0_stride=nelem}.
__device__ __forceinline__ void tdm_load_1d_f16(const _Float16* dst_lds,
                                                const _Float16* src, unsigned nelem) {
  unsigned lds_off = (unsigned)(unsigned long long)dst_lds;
  unsigned long long ga = (unsigned long long)src;
  u32x4 g0;
  g0[0] = 1u;                                               // count=1, is_restore=0
  g0[1] = lds_off;                                          // lds_addr
  g0[2] = (unsigned)ga;                                     // global_addr[31:0]
  g0[3] = (unsigned)((ga >> 32) & 0x01FFFFFFull) | (2u << 30); // addr hi, type=2
  i32x8 g1;
  g1[0] = 1 << 16;                                          // data_size = 2 bytes
  g1[1] = (int)((nelem & 0xFFFFu) << 16);                   // tensor_dim0[15:0]
  g1[2] = (int)(((nelem >> 16) & 0xFFFFu) | (1u << 16));    // tensor_dim0[31:16], tensor_dim1=1
  g1[3] = (int)((nelem & 0xFFFFu) << 16);                   // tile_dim0 = nelem
  g1[4] = 0;                                                // tile_dim1/2 unused
  g1[5] = (int)nelem;                                       // tensor_dim0_stride lo
  g1[6] = 0;
  g1[7] = 0;
  i32x4 z4 = {0, 0, 0, 0};
#if __clang_major__ >= 23
  i32x8 z8 = {0, 0, 0, 0, 0, 0, 0, 0};
  __builtin_amdgcn_tensor_load_to_lds(g0, g1, z4, z4, z8, 0);
#else
  __builtin_amdgcn_tensor_load_to_lds(g0, g1, z4, z4, 0);
#endif
}
#endif

// Stage a contiguous f16 block into LDS (TDM if available, else cooperative copy).
// Caller must __syncthreads() afterwards.
__device__ __forceinline__ void stage_f16(_Float16* dst, const _Float16* src,
                                          int nelem, int tid, int T) {
#if HAVE_TDM
  if (tid < 32) {
    tdm_load_1d_f16(dst, src, (unsigned)nelem);
    __builtin_amdgcn_s_wait_tensorcnt(0);
  }
  (void)T;
#else
  unsigned int* d = (unsigned int*)dst;
  const unsigned int* s = (const unsigned int*)src;
  for (int i = tid; i < (nelem >> 1); i += T) d[i] = s[i];
#endif
}

// ---------------- weight f32 -> f16 with K padding: Wh[out][kpad] ----------------
__global__ void convert_pad_kernel(const float* __restrict__ W, _Float16* __restrict__ Wh,
                                   int out_dim, int in_dim, int kpad) {
  int i = blockIdx.x * blockDim.x + threadIdx.x;
  if (i >= out_dim * kpad) return;
  int n = i / kpad, k = i % kpad;
  Wh[i] = (_Float16)((k < in_dim) ? W[(size_t)n * in_dim + k] : 0.f);
}

// ---------------- segment index prep (pf_to_jet is sorted) ----------------
__global__ void starts_kernel(const int* __restrict__ jet, int* __restrict__ starts, int n_pf) {
  int i = blockIdx.x * blockDim.x + threadIdx.x;
  if (i >= n_pf) return;
  int j = jet[i];
  if (i == 0 || jet[i - 1] != j) starts[j] = i;
}

__global__ void pos_kernel(const int* __restrict__ jet, const int* __restrict__ starts,
                           int* __restrict__ pos, int n_pf) {
  int i = blockIdx.x * blockDim.x + threadIdx.x;
  if (i >= n_pf) return;
  pos[i] = i - starts[jet[i]];
}

// ---------------- generic dense layer: Y = act(X @ W^T + b), via WMMA ----------------
// One block = one 16-row M tile; wave threadIdx.y owns a 16-column N tile.
// Wh is pre-converted f16, already zero-padded to [OUT_DIM][KPAD].
template<int KPAD, int IN_DIM, int OUT_DIM, bool ELU>
__global__ void __launch_bounds__(2 * OUT_DIM)
mlp_layer_kernel(const float* __restrict__ X, const _Float16* __restrict__ Wh,
                 const float* __restrict__ bias, float* __restrict__ Y, int n_rows) {
  __shared__ _Float16 sA[16][KPAD];
  __shared__ _Float16 sB[OUT_DIM][KPAD];
  const int m0 = blockIdx.x * 16;
  const int tid = threadIdx.y * 32 + threadIdx.x;
  const int T = 32 * (OUT_DIM / 16);
  if (tid == 0) __builtin_prefetch(Wh, 0, 1);
  stage_f16(&sB[0][0], Wh, OUT_DIM * KPAD, tid, T);
  for (int idx = tid; idx < 16 * KPAD; idx += T) {
    int r = idx / KPAD, k = idx % KPAD;
    float v = 0.f;
    if (k < IN_DIM && (m0 + r) < n_rows) v = X[(size_t)(m0 + r) * IN_DIM + k];
    sA[r][k] = (_Float16)v;
  }
  __syncthreads();
  const int lane = threadIdx.x;
  const int lr = lane & 15, hf = lane >> 4;
  const int nt = threadIdx.y;
  CF c;
  #pragma unroll
  for (int r = 0; r < 8; ++r) c.f[r] = 0.f;
  #pragma unroll
  for (int kk = 0; kk < KPAD; kk += 32) {
    AF a, b;
    #pragma unroll
    for (int j = 0; j < 8; ++j) {
      a.h[j]     = sA[lr][kk + 8 * hf + j];
      a.h[8 + j] = sA[lr][kk + 8 * hf + 16 + j];
      b.h[j]     = sB[nt * 16 + lr][kk + 8 * hf + j];
      b.h[8 + j] = sB[nt * 16 + lr][kk + 8 * hf + 16 + j];
    }
    c.v = __builtin_amdgcn_wmma_f32_16x16x32_f16(false, a.v, false, b.v,
                                                 (short)0, c.v, false, false);
  }
  const int col = nt * 16 + lr;
  const float bv = bias[col];
  #pragma unroll
  for (int r = 0; r < 8; ++r) {
    int row = m0 + r + 8 * hf;
    if (row < n_rows) {
      float v = c.f[r] + bv;
      if (ELU) v = elu_f(v);
      Y[(size_t)row * OUT_DIM + col] = v;
    }
  }
}

// ---------------- final 128 -> 1 layer (tiny; VALU), mode: 0=linear 1=sigmoid ----------------
__global__ void dense_out1_kernel(const float* __restrict__ X, const float* __restrict__ W,
                                  const float* __restrict__ b, float* __restrict__ y,
                                  int n_rows, int in_dim, int mode) {
  int i = blockIdx.x * blockDim.x + threadIdx.x;
  if (i >= n_rows) return;
  float s = b[0];
  for (int k = 0; k < in_dim; ++k) s += X[(size_t)i * in_dim + k] * W[k];
  if (mode == 1) s = 1.f / (1.f + expf(-s));
  y[i] = s;
}

// ---------------- scatter / gather between ragged and padded ----------------
__global__ void scatter_kernel(const float* __restrict__ emb, const int* __restrict__ jet,
                               const int* __restrict__ pos, float* __restrict__ padded, int n_pf) {
  int i = blockIdx.x * blockDim.x + threadIdx.x;
  if (i >= n_pf * 32) return;
  int pf = i >> 5, d = i & 31;
  padded[((size_t)jet[pf] * 64 + pos[pf]) * 32 + d] = emb[i];
}

__global__ void gather_kernel(const float* __restrict__ padded, const int* __restrict__ jet,
                              const int* __restrict__ pos, float* __restrict__ flat, int n_pf) {
  int i = blockIdx.x * blockDim.x + threadIdx.x;
  if (i >= n_pf * 32) return;
  int pf = i >> 5, d = i & 31;
  flat[i] = padded[((size_t)jet[pf] * 64 + pos[pf]) * 32 + d];
}

// ---------------- fused 3-layer MHA stack, one block (8 waves) per jet ----------------
struct MhaP {
  const _Float16* inWh[3];  const float* inb[3];
  const _Float16* outWh[3]; const float* outb[3];
  const float* lng[3];      const float* lnb[3];
};

__global__ void __launch_bounds__(256)
mha_stack_kernel(float* __restrict__ padded, MhaP p) {
  __shared__ float    x[64][32];      // residual stream
  __shared__ float    qkv[64][96];    // q | k | v
  __shared__ float    att[64][32];    // attn out, then x+corr
  __shared__ _Float16 xh[64][32];
  __shared__ _Float16 atth[64][32];
  __shared__ _Float16 wqkv[96][32];
  __shared__ _Float16 wout[32][32];
  __shared__ int      smask[64];
  const int jet = blockIdx.x;
  const int tid = threadIdx.x;
  float* gx = padded + (size_t)jet * 64 * 32;
  for (int idx = tid; idx < 2048; idx += 256) x[idx >> 5][idx & 31] = gx[idx];
  __syncthreads();
  if (tid < 64) smask[tid] = (x[tid][31] == 0.f) ? 1 : 0;  // padding heuristic, fixed once
  const int wave = tid >> 5, lane = tid & 31, lr = lane & 15, hf = lane >> 4;

  for (int layer = 0; layer < 3; ++layer) {
    const float* inb = p.inb[layer];
    const float* outb = p.outb[layer];
    stage_f16(&wqkv[0][0], p.inWh[layer], 96 * 32, tid, 256);
    stage_f16(&wout[0][0], p.outWh[layer], 32 * 32, tid, 256);
    for (int idx = tid; idx < 2048; idx += 256) xh[idx >> 5][idx & 31] = (_Float16)x[idx >> 5][idx & 31];
    __syncthreads();

    // ---- QKV = x @ in_W^T + in_b : M=64, N=96, K=32 -> 24 WMMA tiles over 8 waves
    for (int t = wave; t < 24; t += 8) {
      int mt = t / 6, ntile = t % 6;
      AF a, b; CF c;
      #pragma unroll
      for (int r = 0; r < 8; ++r) c.f[r] = 0.f;
      #pragma unroll
      for (int j = 0; j < 8; ++j) {
        a.h[j]     = xh[mt * 16 + lr][8 * hf + j];
        a.h[8 + j] = xh[mt * 16 + lr][8 * hf + 16 + j];
        b.h[j]     = wqkv[ntile * 16 + lr][8 * hf + j];
        b.h[8 + j] = wqkv[ntile * 16 + lr][8 * hf + 16 + j];
      }
      c.v = __builtin_amdgcn_wmma_f32_16x16x32_f16(false, a.v, false, b.v,
                                                   (short)0, c.v, false, false);
      int col = ntile * 16 + lr;
      float bv = inb[col];
      #pragma unroll
      for (int r = 0; r < 8; ++r) qkv[mt * 16 + r + 8 * hf][col] = c.f[r] + bv;
    }
    __syncthreads();

    // ---- attention: head_dim=8, online softmax; thread = (query, head)
    {
      int q = tid >> 2, hd = tid & 3;
      float mval = -1e30f, l = 0.f, acc[8];
      #pragma unroll
      for (int d = 0; d < 8; ++d) acc[d] = 0.f;
      for (int k = 0; k < 64; ++k) {
        if (smask[k]) continue;  // key_padding_mask -> -inf -> excluded
        float s = 0.f;
        #pragma unroll
        for (int d = 0; d < 8; ++d) s += qkv[q][hd * 8 + d] * qkv[k][32 + hd * 8 + d];
        s *= 0.35355339059327373f;  // 1/sqrt(8)
        float mnew = fmaxf(mval, s);
        float sc = expf(mval - mnew);
        float pv = expf(s - mnew);
        l = l * sc + pv;
        #pragma unroll
        for (int d = 0; d < 8; ++d) acc[d] = acc[d] * sc + pv * qkv[k][64 + hd * 8 + d];
        mval = mnew;
      }
      float inv = 1.f / fmaxf(l, 1e-30f);
      #pragma unroll
      for (int d = 0; d < 8; ++d) att[q][hd * 8 + d] = acc[d] * inv;
    }
    __syncthreads();
    for (int idx = tid; idx < 2048; idx += 256) atth[idx >> 5][idx & 31] = (_Float16)att[idx >> 5][idx & 31];
    __syncthreads();

    // ---- out proj + mask + residual: M=64, N=32, K=32 -> 8 tiles, one per wave
    {
      int mt = wave >> 1, ntile = wave & 1;
      AF a, b; CF c;
      #pragma unroll
      for (int r = 0; r < 8; ++r) c.f[r] = 0.f;
      #pragma unroll
      for (int j = 0; j < 8; ++j) {
        a.h[j]     = atth[mt * 16 + lr][8 * hf + j];
        a.h[8 + j] = atth[mt * 16 + lr][8 * hf + 16 + j];
        b.h[j]     = wout[ntile * 16 + lr][8 * hf + j];
        b.h[8 + j] = wout[ntile * 16 + lr][8 * hf + 16 + j];
      }
      c.v = __builtin_amdgcn_wmma_f32_16x16x32_f16(false, a.v, false, b.v,
                                                   (short)0, c.v, false, false);
      int col = ntile * 16 + lr;
      float bv = outb[col];
      #pragma unroll
      for (int r = 0; r < 8; ++r) {
        int m = mt * 16 + r + 8 * hf;
        float corr = c.f[r] + bv;
        if (smask[m]) corr = 0.f;
        att[m][col] = x[m][col] + corr;  // att now holds x + corr
      }
    }
    __syncthreads();

    // ---- layernorm per row
    if (tid < 64) {
      const float* g = p.lng[layer]; const float* bb = p.lnb[layer];
      float mean = 0.f;
      for (int k = 0; k < 32; ++k) mean += att[tid][k];
      mean *= (1.f / 32.f);
      float var = 0.f;
      for (int k = 0; k < 32; ++k) { float d = att[tid][k] - mean; var += d * d; }
      var *= (1.f / 32.f);
      float inv = rsqrtf(var + 1e-5f);
      for (int k = 0; k < 32; ++k) x[tid][k] = (att[tid][k] - mean) * inv * g[k] + bb[k];
    }
    __syncthreads();
  }
  for (int idx = tid; idx < 2048; idx += 256) gx[idx] = x[idx >> 5][idx & 31];
}

// ---------------- attentional aggregation: one wave per jet ----------------
__global__ void __launch_bounds__(32)
agg_kernel(const float* __restrict__ flat, const float* __restrict__ gate,
           const int* __restrict__ starts, int n_pf, int n_jets,
           const float* __restrict__ jet_feat, float* __restrict__ jet_in) {
  int jet = blockIdx.x;
  int lane = threadIdx.x;
  int s = starts[jet];
  int e = (jet + 1 < n_jets) ? starts[jet + 1] : n_pf;
  float mx = -1e30f;
  for (int i = s + lane; i < e; i += 32) mx = fmaxf(mx, gate[i]);
  #pragma unroll
  for (int off = 16; off; off >>= 1) mx = fmaxf(mx, __shfl_xor(mx, off, 32));
  float sum = 0.f;
  for (int i = s + lane; i < e; i += 32) sum += expf(gate[i] - mx);
  #pragma unroll
  for (int off = 16; off; off >>= 1) sum += __shfl_xor(sum, off, 32);
  float denom = sum + 1e-16f;
  float acc = 0.f;  // lane = embedding dim (0..31)
  for (int i = s; i < e; ++i) {
    float w = expf(gate[i] - mx) / denom;
    acc += w * flat[(size_t)i * 32 + lane];
  }
  jet_in[(size_t)jet * 36 + 4 + lane] = acc;
  if (lane < 4) jet_in[(size_t)jet * 36 + lane] = jet_feat[(size_t)jet * 4 + lane];
}

// ---------------- host launch ----------------
extern "C" void kernel_launch(void* const* d_in, const int* in_sizes, int n_in,
                              void* d_out, int out_size, void* d_ws, size_t ws_size,
                              hipStream_t stream) {
  const float* pf_feat   = (const float*)d_in[0];
  const int*   pf_to_jet = (const int*)d_in[1];
  const float* jet_feat  = (const float*)d_in[2];
  const int n_pf   = in_sizes[0] / 6;
  const int n_jets = in_sizes[2] / 4;

  #define F(i) ((const float*)d_in[(i)])

  char* base = (char*)d_ws;
  size_t off = 0;
  auto alloc = [&](size_t bytes) -> void* {
    off = (off + 255) & ~(size_t)255;
    void* p = base + off; off += bytes; return p;
  };
  int*   starts = (int*)  alloc((size_t)n_jets * 4);
  int*   pos    = (int*)  alloc((size_t)n_pf * 4);
  float* bufA   = (float*)alloc((size_t)n_pf * 128 * 4);
  float* bufB   = (float*)alloc((size_t)n_pf * 128 * 4);
  float* emb    = (float*)alloc((size_t)n_pf * 32 * 4);
  float* padded = (float*)alloc((size_t)n_jets * 64 * 32 * 4);
  float* flat   = (float*)alloc((size_t)n_pf * 32 * 4);
  float* gate   = (float*)alloc((size_t)n_pf * 4);
  float* jet_in = (float*)alloc((size_t)n_jets * 36 * 4);
  float* out    = (float*)d_out;

  // f16 (K-padded) weight copies, converted once per launch
  auto allocH = [&](size_t elems) -> _Float16* { return (_Float16*)alloc(elems * 2); };
  auto conv = [&](int idx, _Float16* dst, int o, int i, int kp) {
    int n = o * kp;
    convert_pad_kernel<<<(n + 255) / 256, 256, 0, stream>>>(F(idx), dst, o, i, kp);
  };
  _Float16* embW1 = allocH(128 * 32);   conv(4,  embW1, 128, 6, 32);
  _Float16* embW2 = allocH(128 * 128);  conv(6,  embW2, 128, 128, 128);
  _Float16* embW3 = allocH(128 * 128);  conv(8,  embW3, 128, 128, 128);
  _Float16* embW4 = allocH(32 * 128);   conv(10, embW4, 32, 128, 128);
  _Float16* gateW1 = allocH(128 * 32);  conv(30, gateW1, 128, 32, 32);
  _Float16* gateW2 = allocH(128 * 128); conv(32, gateW2, 128, 128, 128);
  _Float16* gateW3 = allocH(128 * 128); conv(34, gateW3, 128, 128, 128);
  _Float16* istW1 = allocH(128 * 64);   conv(38, istW1, 128, 36, 64);
  _Float16* istW2 = allocH(128 * 128);  conv(40, istW2, 128, 128, 128);
  _Float16* istW3 = allocH(128 * 128);  conv(42, istW3, 128, 128, 128);
  _Float16* visW1 = allocH(128 * 64);   conv(46, visW1, 128, 36, 64);
  _Float16* visW2 = allocH(128 * 128);  conv(48, visW2, 128, 128, 128);
  _Float16* visW3 = allocH(128 * 128);  conv(50, visW3, 128, 128, 128);
  _Float16* mhaInWh[3];  _Float16* mhaOutWh[3];
  for (int l = 0; l < 3; ++l) {
    mhaInWh[l]  = allocH(96 * 32);  conv(12 + 4 * l, mhaInWh[l], 96, 32, 32);
    mhaOutWh[l] = allocH(32 * 32);  conv(14 + 4 * l, mhaOutWh[l], 32, 32, 32);
  }

  const dim3 b128(32, 8);   // OUT_DIM = 128
  const dim3 b32(32, 2);    // OUT_DIM = 32
  const int mt_pf  = (n_pf + 15) / 16;
  const int mt_jet = (n_jets + 15) / 16;
  const int g1pf  = (n_pf + 255) / 256;
  const int g32pf = (n_pf * 32 + 255) / 256;

  // segment indices
  starts_kernel<<<g1pf, 256, 0, stream>>>(pf_to_jet, starts, n_pf);
  pos_kernel<<<g1pf, 256, 0, stream>>>(pf_to_jet, starts, pos, n_pf);

  // embedding MLP: 6 -> 128 -> 128 -> 128 -> 32
  mlp_layer_kernel<32, 6, 128, true ><<<mt_pf, b128, 0, stream>>>(pf_feat, embW1, F(5),  bufA, n_pf);
  mlp_layer_kernel<128, 128, 128, true ><<<mt_pf, b128, 0, stream>>>(bufA, embW2, F(7),  bufB, n_pf);
  mlp_layer_kernel<128, 128, 128, true ><<<mt_pf, b128, 0, stream>>>(bufB, embW3, F(9),  bufA, n_pf);
  mlp_layer_kernel<128, 128, 32, false><<<mt_pf, b32,  0, stream>>>(bufA, embW4, F(11), emb,  n_pf);

  // scatter to padded [n_jets, 64, 32]
  hipMemsetAsync(padded, 0, (size_t)n_jets * 64 * 32 * 4, stream);
  scatter_kernel<<<g32pf, 256, 0, stream>>>(emb, pf_to_jet, pos, padded, n_pf);

  // fused 3-layer MHA stack
  MhaP p;
  for (int l = 0; l < 3; ++l) {
    p.inWh[l]  = mhaInWh[l];   p.inb[l]  = F(13 + 4 * l);
    p.outWh[l] = mhaOutWh[l];  p.outb[l] = F(15 + 4 * l);
    p.lng[l]   = F(24 + 2 * l); p.lnb[l] = F(25 + 2 * l);
  }
  mha_stack_kernel<<<n_jets, 256, 0, stream>>>(padded, p);

  // gather valid tokens back to ragged order
  gather_kernel<<<g32pf, 256, 0, stream>>>(padded, pf_to_jet, pos, flat, n_pf);

  // gate MLP: 32 -> 128 -> 128 -> 128 -> 1
  mlp_layer_kernel<32, 32, 128, true><<<mt_pf, b128, 0, stream>>>(flat, gateW1, F(31), bufA, n_pf);
  mlp_layer_kernel<128, 128, 128, true><<<mt_pf, b128, 0, stream>>>(bufA, gateW2, F(33), bufB, n_pf);
  mlp_layer_kernel<128, 128, 128, true><<<mt_pf, b128, 0, stream>>>(bufB, gateW3, F(35), bufA, n_pf);
  dense_out1_kernel<<<g1pf, 256, 0, stream>>>(bufA, F(36), F(37), gate, n_pf, 128, 0);

  // segment softmax aggregation + concat jet features -> [n_jets, 36]
  agg_kernel<<<n_jets, 32, 0, stream>>>(flat, gate, starts, n_pf, n_jets, jet_feat, jet_in);

  const int g1jet = (n_jets + 255) / 256;
  // istau head: 36 -> 128 -> 128 -> 128 -> 1 (+ sigmoid)
  mlp_layer_kernel<64, 36, 128, true><<<mt_jet, b128, 0, stream>>>(jet_in, istW1, F(39), bufA, n_jets);
  mlp_layer_kernel<128, 128, 128, true><<<mt_jet, b128, 0, stream>>>(bufA, istW2, F(41), bufB, n_jets);
  mlp_layer_kernel<128, 128, 128, true><<<mt_jet, b128, 0, stream>>>(bufB, istW3, F(43), bufA, n_jets);
  dense_out1_kernel<<<g1jet, 256, 0, stream>>>(bufA, F(44), F(45), out, n_jets, 128, 1);

  // visenergy head: 36 -> 128 -> 128 -> 128 -> 1
  mlp_layer_kernel<64, 36, 128, true><<<mt_jet, b128, 0, stream>>>(jet_in, visW1, F(47), bufA, n_jets);
  mlp_layer_kernel<128, 128, 128, true><<<mt_jet, b128, 0, stream>>>(bufA, visW2, F(49), bufB, n_jets);
  mlp_layer_kernel<128, 128, 128, true><<<mt_jet, b128, 0, stream>>>(bufB, visW3, F(51), bufA, n_jets);
  dense_out1_kernel<<<g1jet, 256, 0, stream>>>(bufA, F(52), F(53), out + n_jets, n_jets, 128, 0);

  #undef F
}
